// METAModel_32444182954036
// MI455X (gfx1250) — matrix operand order repaired and checked by reading, not compiled, not run on hardware
//
#include <hip/hip_runtime.h>
#include <hip/hip_bf16.h>

// ---------------------------------------------------------------------------
// Types for WMMA fragments (wave32, gfx1250)
// ---------------------------------------------------------------------------
typedef __attribute__((ext_vector_type(16))) _Float16 v16h;
typedef __attribute__((ext_vector_type(8)))  _Float16 v8h;
typedef __attribute__((ext_vector_type(8)))  float    v8f;

#define D_DIM 128
#define H_DIM 8
#define DK 16
#define NNODE 50000
#define NSRC  100000
#define ENBR  800000
#define EINC  200000
#define MITER 5               // M-tiles per wave strip; 3125 = 5*625, 6250 = 5*1250
#define CDIV(a, b) (((a) + (b) - 1) / (b))

// ---------------------------------------------------------------------------
// Small helpers
// ---------------------------------------------------------------------------
__device__ __forceinline__ float sigm(float x) { return 1.0f / (1.0f + expf(-x)); }

// Monotonic float<->uint mapping so atomicMax(uint) == float max.
__device__ __forceinline__ unsigned ford(float f) {
  unsigned u = __float_as_uint(f);
  return (u >> 31) ? ~u : (u | 0x80000000u);
}
__device__ __forceinline__ float orddec(unsigned u) {
  return (u >> 31) ? __uint_as_float(u & 0x7fffffffu) : __uint_as_float(~u);
}

// ---------------------------------------------------------------------------
// Utility kernels
// ---------------------------------------------------------------------------
__global__ void cvt_f16_kernel(const float* __restrict__ src,
                               _Float16* __restrict__ dst, int n) {
  int i = blockIdx.x * blockDim.x + threadIdx.x;
  if (i < n) dst[i] = (_Float16)src[i];
}

__global__ void zero_kernel(unsigned* __restrict__ p, int n) {
  int i = blockIdx.x * blockDim.x + threadIdx.x;
  if (i < n) p[i] = 0u;
}

// ---------------------------------------------------------------------------
// Pack an f32 weight matrix (128 x ldw, 128-column window) into the contiguous-k
// wave32 WMMA B-fragment order, fused with the f32->f16 convert:
//   Bp[(((tn*4+ks)*32+lane)*16)+i] = W[(ks*32 + (lane>=16?16:0) + i)*ldw + tn*16+(lane&15)]
// The GEMM then reads each fragment as two contiguous b128 loads per lane.
// ---------------------------------------------------------------------------
__global__ void pack_b_kernel(const float* __restrict__ W, int ldw,
                              _Float16* __restrict__ Bp) {
  int t = blockIdx.x * blockDim.x + threadIdx.x;
  if (t >= 8 * 2048) return;             // 8 col tiles * 4 ksteps * 32 lanes * 16
  int i    = t & 15;
  int lane = (t >> 4) & 31;
  int ks   = (t >> 9) & 3;
  int tn   = t >> 11;
  int col  = tn * 16 + (lane & 15);
  int k    = ks * 32 + ((lane >= 16) ? 16 : 0) + i;
  Bp[t] = (_Float16)W[(size_t)k * ldw + col];
}

// ---------------------------------------------------------------------------
// WMMA GEMM: C[M x 128] = A[M x 128] * B[128 x 128], A f16 row-major, B in
// packed fragment order, C f32 (ldc = 128 hardcoded -> every access is one
// base register + compile-time immediate offset). One wave owns one
// 16-column tile and a strip of exactly MITER 16-row tiles, so there is NO
// control flow at all: grid is exact, strips are always full, EXEC is
// all-ones by construction at every v_wmma. B fragments (4 K-steps, 32
// VGPRs) are loaded once per wave; each M-tile costs 8 clause-grouped b128
// A-loads + 4 back-to-back WMMAs + 8 immediate-offset stores.
// ---------------------------------------------------------------------------
__global__ void gemm_wmma_kernel(const _Float16* __restrict__ A,
                                 const _Float16* __restrict__ Bp,
                                 float* __restrict__ C) {
  int wave = threadIdx.x >> 5;
  int lane = threadIdx.x & 31;
  int sid  = blockIdx.x * 4 + wave;      // grid sized exactly: no guard
  int strip = sid >> 3;                  // 8 column tiles (N = 128)
  int tn    = sid & 7;
  int hi    = (lane >= 16) ? 8 : 0;      // upper-lane C row shift; A K shift is 2*hi
  int colL  = lane & 15;

  // B fragments for this column tile: 4 K-steps, 2 x b128 each, kept live.
  const _Float16* bbase = Bp + ((size_t)tn * 128 + lane) * 16;
  v16h bv[4];
#pragma unroll
  for (int ks = 0; ks < 4; ++ks) {
    v8h lo  = *(const v8h*)(bbase + ks * 512);
    v8h hii = *(const v8h*)(bbase + ks * 512 + 8);
#pragma unroll
    for (int i = 0; i < 8; ++i) { bv[ks][i] = lo[i]; bv[ks][8 + i] = hii[i]; }
  }

  int m0 = strip * MITER;
  const _Float16* abase = A + ((size_t)m0 * 16 + colL) * 128;
  float* cbase = C + ((size_t)m0 * 16 + hi) * 128 + tn * 16 + colL;

#pragma unroll
  for (int it = 0; it < MITER; ++it) {
    const _Float16* arow = abase + it * 2048;
    // All 8 A-fragment loads for this tile issue as one clause.
    // A fragment 16x32 f16: lanes 0-15 hold K {kk..kk+15},
    // lanes 16-31 hold K {kk+16..kk+31} (matches packed B).
    v8h alo[4], ahi[4];
#pragma unroll
    for (int ks = 0; ks < 4; ++ks) {
      alo[ks] = *(const v8h*)(arow + ks * 32 + 2 * hi);
      ahi[ks] = *(const v8h*)(arow + ks * 32 + 8 + 2 * hi);
    }
    v8f acc = {};
#pragma unroll
    for (int ks = 0; ks < 4; ++ks) {
      v16h av;
#pragma unroll
      for (int i = 0; i < 8; ++i) { av[i] = alo[ks][i]; av[8 + i] = ahi[ks][i]; }
      acc = __builtin_amdgcn_wmma_f32_16x16x32_f16(
          /*neg_a=*/false, av, /*neg_b=*/false, bv[ks],
          /*c_mod=*/(short)0, acc, /*reuse_a=*/false, /*reuse_b=*/false);
    }
    // C 16x16 f32 layout: VGPR j -> row j (lanes 0-15) / row j+8 (lanes 16-31).
#pragma unroll
    for (int j = 0; j < 8; ++j)
      cbase[(size_t)it * 2048 + j * 128] = acc[j];
  }
}

// ---------------------------------------------------------------------------
// Narrow GEMM (N = 8 head-bias projections): one thread per row, f32.
// ---------------------------------------------------------------------------
__global__ void gemm8_kernel(const float* __restrict__ A,
                             const float* __restrict__ W,  // 128 x 8
                             float* __restrict__ C, int M) {
  int r = blockIdx.x * blockDim.x + threadIdx.x;
  if (r >= M) return;
  float acc[8] = {0.f, 0.f, 0.f, 0.f, 0.f, 0.f, 0.f, 0.f};
  const float* a = A + (size_t)r * 128;
  for (int k = 0; k < 128; ++k) {
    float av = a[k];
#pragma unroll
    for (int h = 0; h < 8; ++h) acc[h] += av * W[k * 8 + h];
  }
#pragma unroll
  for (int h = 0; h < 8; ++h) C[(size_t)r * 8 + h] = acc[h];
}

// ---------------------------------------------------------------------------
// Edge pipeline (segmented softmax via atomics; Q/K/V fit in the 192MB L2)
// ---------------------------------------------------------------------------
__global__ void edge_scores_kernel(const float* __restrict__ Q,
                                   const float* __restrict__ K,
                                   const float* __restrict__ biasQ,  // per seg node
                                   const float* __restrict__ biasK,  // may be null
                                   const int* __restrict__ eseg,
                                   const int* __restrict__ eoth,
                                   float* __restrict__ scores,
                                   unsigned* __restrict__ nmax,
                                   int nedge, float scale) {
  int t = blockIdx.x * blockDim.x + threadIdx.x;
  if (t >= nedge * H_DIM) return;
  int e = t >> 3, h = t & 7;
  int s = eseg[e], o = eoth[e];
  const float4* q4 = (const float4*)(Q + (size_t)s * 128 + h * DK);
  const float4* k4 = (const float4*)(K + (size_t)o * 128 + h * DK);
  float acc = 0.f;
#pragma unroll
  for (int i = 0; i < 4; ++i) {
    float4 qa = q4[i], ka = k4[i];
    acc += qa.x * ka.x + qa.y * ka.y + qa.z * ka.z + qa.w * ka.w;
  }
  float sc = acc * scale + biasQ[(size_t)s * 8 + h];
  if (biasK) sc += biasK[(size_t)o * 8 + h];
  scores[(size_t)e * 8 + h] = sc;
  atomicMax(&nmax[(size_t)s * 8 + h], ford(sc));
}

__global__ void edge_exp_kernel(float* __restrict__ scores,
                                const unsigned* __restrict__ nmax,
                                float* __restrict__ denom,
                                const int* __restrict__ eseg, int nedge) {
  int t = blockIdx.x * blockDim.x + threadIdx.x;
  if (t >= nedge * H_DIM) return;
  int e = t >> 3, h = t & 7;
  int s = eseg[e];
  float mx = orddec(nmax[(size_t)s * 8 + h]);
  float ex = expf(scores[(size_t)e * 8 + h] - mx);
  scores[(size_t)e * 8 + h] = ex;
  atomicAdd(&denom[(size_t)s * 8 + h], ex);
}

__global__ void edge_accum_kernel(const float* __restrict__ ex,
                                  const float* __restrict__ V,
                                  const int* __restrict__ eseg,
                                  const int* __restrict__ eoth,
                                  float* __restrict__ outacc, int nedge) {
  int t = blockIdx.x * blockDim.x + threadIdx.x;
  if (t >= nedge * D_DIM) return;
  int e = t >> 7, d = t & 127, h = d >> 4;
  int s = eseg[e], o = eoth[e];
  atomicAdd(&outacc[(size_t)s * 128 + d],
            ex[(size_t)e * 8 + h] * V[(size_t)o * 128 + d]);
}

// ---------------------------------------------------------------------------
// Gating / normalization fusions
// ---------------------------------------------------------------------------
__global__ void gate_mul_norm_kernel(const float* __restrict__ acc,
                                     const float* __restrict__ denom,
                                     const float* __restrict__ glin,
                                     const float* __restrict__ gb,
                                     _Float16* __restrict__ outh, int M) {
  int t = blockIdx.x * blockDim.x + threadIdx.x;
  if (t >= M * D_DIM) return;
  int n = t >> 7, d = t & 127, h = d >> 4;
  float g = sigm(glin[t] + gb[d]);
  float v = acc[t] / (denom[(size_t)n * 8 + h] + 1e-12f);
  outh[t] = (_Float16)(g * v);
}

__global__ void gate_mul_plain_kernel(const float* __restrict__ glin,
                                      const float* __restrict__ gb,
                                      const float* __restrict__ V,
                                      float* __restrict__ out, int M) {
  int t = blockIdx.x * blockDim.x + threadIdx.x;
  if (t >= M * D_DIM) return;
  int d = t & 127;
  out[t] = sigm(glin[t] + gb[d]) * V[t];
}

// ---------------------------------------------------------------------------
// LayerNorm(x + y) -> f32 (and optional f16 copy). One wave per row.
// ---------------------------------------------------------------------------
__global__ void ln_res_kernel(const float* __restrict__ x,
                              const float* __restrict__ y,
                              const float* __restrict__ g,
                              const float* __restrict__ b,
                              float* __restrict__ o32,
                              _Float16* __restrict__ o16, int M) {
  int wave = threadIdx.x >> 5;
  int lane = threadIdx.x & 31;
  int row = blockIdx.x * 4 + wave;
  if (row >= M) return;
  float v[4];
  float s = 0.f;
#pragma unroll
  for (int t = 0; t < 4; ++t) {
    int idx = lane + 32 * t;
    v[t] = x[(size_t)row * 128 + idx] + y[(size_t)row * 128 + idx];
    s += v[t];
  }
#pragma unroll
  for (int o = 16; o > 0; o >>= 1) s += __shfl_xor(s, o, 32);
  float mean = s * (1.0f / 128.0f);
  float q = 0.f;
#pragma unroll
  for (int t = 0; t < 4; ++t) { float d = v[t] - mean; q += d * d; }
#pragma unroll
  for (int o = 16; o > 0; o >>= 1) q += __shfl_xor(q, o, 32);
  float inv = rsqrtf(q * (1.0f / 128.0f) + 1e-5f);
#pragma unroll
  for (int t = 0; t < 4; ++t) {
    int idx = lane + 32 * t;
    float r = g[idx] * (v[t] - mean) * inv + b[idx];
    if (o32) o32[(size_t)row * 128 + idx] = r;
    if (o16) o16[(size_t)row * 128 + idx] = (_Float16)r;
  }
}

// ---------------------------------------------------------------------------
// Host orchestration
// ---------------------------------------------------------------------------
extern "C" void kernel_launch(void* const* d_in, const int* in_sizes, int n_in,
                              void* d_out, int out_size, void* d_ws, size_t ws_size,
                              hipStream_t stream) {
  (void)in_sizes; (void)n_in; (void)out_size; (void)ws_size;
  const float* X         = (const float*)d_in[0];
  const float* X_src     = (const float*)d_in[1];
  const float* W_qkv     = (const float*)d_in[2];
  const float* w_bias    = (const float*)d_in[3];
  const float* W_gate    = (const float*)d_in[4];
  const float* b_gate    = (const float*)d_in[5];
  const float* W_o       = (const float*)d_in[6];
  const float* W_q       = (const float*)d_in[7];
  const float* W_kv      = (const float*)d_in[8];
  const float* w_bias_t  = (const float*)d_in[9];
  const float* w_bias_s  = (const float*)d_in[10];
  const float* W_gate_t  = (const float*)d_in[11];
  const float* b_gate_t  = (const float*)d_in[12];
  const float* W_gate_s  = (const float*)d_in[13];
  const float* b_gate_s  = (const float*)d_in[14];
  const float* W_o2      = (const float*)d_in[15];
  const float* ln1_g     = (const float*)d_in[16];
  const float* ln1_b     = (const float*)d_in[17];
  const float* ln2_g     = (const float*)d_in[18];
  const float* ln2_b     = (const float*)d_in[19];
  const int*   nbr_src   = (const int*)d_in[20];
  const int*   nbr_dst   = (const int*)d_in[21];
  const int*   inc_tgt   = (const int*)d_in[22];
  const int*   inc_src   = (const int*)d_in[23];

  char* ws = (char*)d_ws;
  size_t off = 0;
  auto alloc = [&](size_t bytes) -> void* {
    void* p = ws + off;
    off = (off + bytes + 255) & ~(size_t)255;
    return p;
  };

  // f16 activations + packed weight fragments (128x128 each)
  _Float16* Xh   = (_Float16*)alloc((size_t)NNODE * 128 * 2);
  _Float16* Xsh  = (_Float16*)alloc((size_t)NSRC * 128 * 2);
  _Float16* BpQ  = (_Float16*)alloc(128 * 128 * 2);
  _Float16* BpK  = (_Float16*)alloc(128 * 128 * 2);
  _Float16* BpV  = (_Float16*)alloc(128 * 128 * 2);
  _Float16* BpG  = (_Float16*)alloc(128 * 128 * 2);
  _Float16* BpO  = (_Float16*)alloc(128 * 128 * 2);
  _Float16* BpQ2 = (_Float16*)alloc(128 * 128 * 2);
  _Float16* BpKc = (_Float16*)alloc(128 * 128 * 2);
  _Float16* BpVc = (_Float16*)alloc(128 * 128 * 2);
  _Float16* BpGs = (_Float16*)alloc(128 * 128 * 2);
  _Float16* BpGt = (_Float16*)alloc(128 * 128 * 2);
  _Float16* BpO2 = (_Float16*)alloc(128 * 128 * 2);
  // stage 1 scratch
  float* Q        = (float*)alloc((size_t)NNODE * 128 * 4);
  float* Kb       = (float*)alloc((size_t)NNODE * 128 * 4);
  float* V        = (float*)alloc((size_t)NNODE * 128 * 4);
  float* gate_lin = (float*)alloc((size_t)NNODE * 128 * 4);
  float* bias_n   = (float*)alloc((size_t)NNODE * 8 * 4);
  float* sc1      = (float*)alloc((size_t)ENBR * 8 * 4);
  unsigned* nmax1 = (unsigned*)alloc((size_t)NNODE * 8 * 4);
  float* den1     = (float*)alloc((size_t)NNODE * 8 * 4);
  float* acc1     = (float*)alloc((size_t)NNODE * 128 * 4);
  _Float16* gatedh = (_Float16*)alloc((size_t)NNODE * 128 * 2);
  float* sa       = (float*)alloc((size_t)NNODE * 128 * 4);
  float* hbuf     = (float*)alloc((size_t)NNODE * 128 * 4);
  _Float16* hh    = (_Float16*)alloc((size_t)NNODE * 128 * 2);
  // stage 2 scratch
  float* Qc       = (float*)alloc((size_t)NNODE * 128 * 4);
  float* Kc       = (float*)alloc((size_t)NSRC * 128 * 4);
  float* Vc       = (float*)alloc((size_t)NSRC * 128 * 4);
  float* gs_lin   = (float*)alloc((size_t)NSRC * 128 * 4);
  float* gt_lin   = (float*)alloc((size_t)NNODE * 128 * 4);
  float* bias_tg  = (float*)alloc((size_t)NNODE * 8 * 4);
  float* bias_sr  = (float*)alloc((size_t)NSRC * 8 * 4);
  float* gV       = (float*)alloc((size_t)NSRC * 128 * 4);
  float* sc2      = (float*)alloc((size_t)EINC * 8 * 4);
  unsigned* nmax2 = (unsigned*)alloc((size_t)NNODE * 8 * 4);
  float* den2     = (float*)alloc((size_t)NNODE * 8 * 4);
  float* acc2     = (float*)alloc((size_t)NNODE * 128 * 4);
  _Float16* gated2h = (_Float16*)alloc((size_t)NNODE * 128 * 2);
  float* ca       = (float*)alloc((size_t)NNODE * 128 * 4);

  const int TB = 256;
  const float scale = 0.25f;  // 1/sqrt(dk=16)
  const int ST_N = (NNODE / 16) / MITER;    // 625 full strips (nodes)
  const int ST_S = (NSRC / 16) / MITER;     // 1250 full strips (sources)
  const int GB_N = ST_N * 2;                // exact grid: ST*8 waves / 4 per block
  const int GB_S = ST_S * 2;
  const int PKB = CDIV(8 * 2048, TB);

  // ---- activation converts + weight fragment packing ----
  cvt_f16_kernel<<<CDIV(NNODE * 128, TB), TB, 0, stream>>>(X, Xh, NNODE * 128);
  cvt_f16_kernel<<<CDIV(NSRC * 128, TB), TB, 0, stream>>>(X_src, Xsh, NSRC * 128);
  pack_b_kernel<<<PKB, TB, 0, stream>>>(W_qkv + 0,   384, BpQ);
  pack_b_kernel<<<PKB, TB, 0, stream>>>(W_qkv + 128, 384, BpK);
  pack_b_kernel<<<PKB, TB, 0, stream>>>(W_qkv + 256, 384, BpV);
  pack_b_kernel<<<PKB, TB, 0, stream>>>(W_gate,   128, BpG);
  pack_b_kernel<<<PKB, TB, 0, stream>>>(W_o,      128, BpO);
  pack_b_kernel<<<PKB, TB, 0, stream>>>(W_q,      128, BpQ2);
  pack_b_kernel<<<PKB, TB, 0, stream>>>(W_kv + 0,   256, BpKc);
  pack_b_kernel<<<PKB, TB, 0, stream>>>(W_kv + 128, 256, BpVc);
  pack_b_kernel<<<PKB, TB, 0, stream>>>(W_gate_s, 128, BpGs);
  pack_b_kernel<<<PKB, TB, 0, stream>>>(W_gate_t, 128, BpGt);
  pack_b_kernel<<<PKB, TB, 0, stream>>>(W_o2,     128, BpO2);

  // ---- zero accumulators (every call; graph replay safe) ----
  zero_kernel<<<CDIV(NNODE * 8, TB), TB, 0, stream>>>(nmax1, NNODE * 8);
  zero_kernel<<<CDIV(NNODE * 8, TB), TB, 0, stream>>>((unsigned*)den1, NNODE * 8);
  zero_kernel<<<CDIV(NNODE * 128, TB), TB, 0, stream>>>((unsigned*)acc1, NNODE * 128);
  zero_kernel<<<CDIV(NNODE * 8, TB), TB, 0, stream>>>(nmax2, NNODE * 8);
  zero_kernel<<<CDIV(NNODE * 8, TB), TB, 0, stream>>>((unsigned*)den2, NNODE * 8);
  zero_kernel<<<CDIV(NNODE * 128, TB), TB, 0, stream>>>((unsigned*)acc2, NNODE * 128);

  // ---- stage 1: self attention over neighbor edges ----
  gemm_wmma_kernel<<<GB_N, 128, 0, stream>>>(Xh, BpQ, Q);
  gemm_wmma_kernel<<<GB_N, 128, 0, stream>>>(Xh, BpK, Kb);
  gemm_wmma_kernel<<<GB_N, 128, 0, stream>>>(Xh, BpV, V);
  gemm_wmma_kernel<<<GB_N, 128, 0, stream>>>(Xh, BpG, gate_lin);
  gemm8_kernel<<<CDIV(NNODE, TB), TB, 0, stream>>>(X, w_bias, bias_n, NNODE);

  edge_scores_kernel<<<CDIV(ENBR * 8, TB), TB, 0, stream>>>(
      Q, Kb, bias_n, nullptr, nbr_src, nbr_dst, sc1, nmax1, ENBR, scale);
  edge_exp_kernel<<<CDIV(ENBR * 8, TB), TB, 0, stream>>>(sc1, nmax1, den1, nbr_src, ENBR);
  edge_accum_kernel<<<CDIV(ENBR * 128, TB), TB, 0, stream>>>(sc1, V, nbr_src, nbr_dst, acc1, ENBR);

  gate_mul_norm_kernel<<<CDIV(NNODE * 128, TB), TB, 0, stream>>>(
      acc1, den1, gate_lin, b_gate, gatedh, NNODE);
  gemm_wmma_kernel<<<GB_N, 128, 0, stream>>>(gatedh, BpO, sa);
  ln_res_kernel<<<CDIV(NNODE, 4), 128, 0, stream>>>(X, sa, ln1_g, ln1_b, hbuf, hh, NNODE);

  // ---- stage 2: cross attention over incidence edges ----
  gemm_wmma_kernel<<<GB_N, 128, 0, stream>>>(hh, BpQ2, Qc);
  gemm_wmma_kernel<<<GB_S, 128, 0, stream>>>(Xsh, BpKc, Kc);
  gemm_wmma_kernel<<<GB_S, 128, 0, stream>>>(Xsh, BpVc, Vc);
  gemm_wmma_kernel<<<GB_S, 128, 0, stream>>>(Xsh, BpGs, gs_lin);
  gemm_wmma_kernel<<<GB_N, 128, 0, stream>>>(hh, BpGt, gt_lin);
  gemm8_kernel<<<CDIV(NNODE, TB), TB, 0, stream>>>(hbuf, w_bias_t, bias_tg, NNODE);
  gemm8_kernel<<<CDIV(NSRC, TB), TB, 0, stream>>>(X_src, w_bias_s, bias_sr, NSRC);
  gate_mul_plain_kernel<<<CDIV(NSRC * 128, TB), TB, 0, stream>>>(gs_lin, b_gate_s, Vc, gV, NSRC);

  edge_scores_kernel<<<CDIV(EINC * 8, TB), TB, 0, stream>>>(
      Qc, Kc, bias_tg, bias_sr, inc_tgt, inc_src, sc2, nmax2, EINC, scale);
  edge_exp_kernel<<<CDIV(EINC * 8, TB), TB, 0, stream>>>(sc2, nmax2, den2, inc_tgt, EINC);
  edge_accum_kernel<<<CDIV(EINC * 128, TB), TB, 0, stream>>>(sc2, gV, inc_tgt, inc_src, acc2, EINC);

  gate_mul_norm_kernel<<<CDIV(NNODE * 128, TB), TB, 0, stream>>>(
      acc2, den2, gt_lin, b_gate_t, gated2h, NNODE);
  gemm_wmma_kernel<<<GB_N, 128, 0, stream>>>(gated2h, BpO2, ca);
  ln_res_kernel<<<CDIV(NNODE, 4), 128, 0, stream>>>(hbuf, ca, ln2_g, ln2_b,
                                                    (float*)d_out, nullptr, NNODE);
}
